// Attention_30107720745394
// MI455X (gfx1250) — matrix operand order repaired and checked by reading, not compiled, not run on hardware
//
#include <hip/hip_runtime.h>

#define BB   2
#define SS   2048
#define DD   1024
#define HH   16
#define HKV  4
#define HD   64
#define REP  (HH / HKV)
#define NQK  (HH * HD)              // 1024 q-proj cols
#define NTOT (NQK + 2 * HKV * HD)   // 1536 = q|k|v concat
#define SCALEF 0.125f               // 1/sqrt(64)
#define NEGV  -1000000000.0f

typedef __attribute__((ext_vector_type(16))) __bf16   v16bf;
typedef __attribute__((ext_vector_type(8)))  float    v8f;
typedef __attribute__((ext_vector_type(4)))  float    v4f;
typedef __attribute__((ext_vector_type(4)))  unsigned v4u;
typedef __attribute__((ext_vector_type(2)))  unsigned v2u;
typedef __attribute__((ext_vector_type(2)))  int      v2i;

#define GLOBAL_AS __attribute__((address_space(1)))
#define LDS_AS    __attribute__((address_space(3)))

union BF16x16 { v16bf v; unsigned u[8]; v4u q[2]; };

#if defined(__has_builtin)
#  define HAS_PK_BF16    __has_builtin(__builtin_amdgcn_cvt_pk_bf16_f32)
#  define HAS_ASYNC_LDS  __has_builtin(__builtin_amdgcn_global_load_async_to_lds_b64)
#  define HAS_WAIT_ASYNC __has_builtin(__builtin_amdgcn_s_wait_asynccnt)
#else
#  define HAS_PK_BF16 0
#  define HAS_ASYNC_LDS 0
#  define HAS_WAIT_ASYNC 0
#endif

__device__ inline unsigned short f2bf(float f) {
  unsigned u = __float_as_uint(f);
  u += 0x7FFFu + ((u >> 16) & 1u);   // round-to-nearest-even
  return (unsigned short)(u >> 16);
}

#if HAS_PK_BF16
typedef __attribute__((ext_vector_type(2))) __bf16 v2bf;
__device__ inline unsigned pack2f(float lo, float hi) {
  union { v2bf v; unsigned u; } r;
  r.v = __builtin_amdgcn_cvt_pk_bf16_f32(lo, hi);
  return r.u;
}
#else
__device__ inline unsigned pack2f(float lo, float hi) {
  return (unsigned)f2bf(lo) | ((unsigned)f2bf(hi) << 16);
}
#endif

__device__ inline v8f wmma_bf16(const BF16x16& a, const BF16x16& b, v8f c) {
  return __builtin_amdgcn_wmma_f32_16x16x32_bf16(
      false, a.v, false, b.v, (short)0, c, false, false);
}

#if HAS_ASYNC_LDS
__device__ inline void wait_async0() {
#if HAS_WAIT_ASYNC
  __builtin_amdgcn_s_wait_asynccnt(0);
#else
  asm volatile("s_wait_asynccnt 0x0" ::: "memory");
#endif
}
#endif

// ---------------------------------------------------------------------------
// Kernel 0: one-shot f32 -> bf16 conversion (bandwidth-bound).
// Removes all conversion VALU from the GEMM inner loops.
// ---------------------------------------------------------------------------
__global__ __launch_bounds__(256) void cvt_bf16_kernel(
    const float* __restrict__ src, unsigned short* __restrict__ dst, int n4)
{
  int i = blockIdx.x * blockDim.x + threadIdx.x;
  if (i < n4) {
    v4f f = *(const v4f*)(src + (size_t)i * 4);
    v2u p = { pack2f(f[0], f[1]), pack2f(f[2], f[3]) };
    *(v2u*)(dst + (size_t)i * 4) = p;
  }
}

// ---------------------------------------------------------------------------
// Kernel 1: QKV projection + RoPE from pre-converted bf16 x and weights.
// One wave = one 16(M) x 16(N) tile of the concat [q|k|v] output (N=1536).
// x tile (bf16, 1KB) staged in LDS via async-to-LDS. V stored TRANSPOSED.
// ---------------------------------------------------------------------------
__global__ __launch_bounds__(256) void qkv_rope_kernel(
    const unsigned short* __restrict__ xbf,
    const unsigned short* __restrict__ wqbf,
    const unsigned short* __restrict__ wkbf,
    const unsigned short* __restrict__ wvbf,
    const float* __restrict__ fcos, const float* __restrict__ fsin,
    unsigned short* __restrict__ qbf, unsigned short* __restrict__ kbf,
    unsigned short* __restrict__ vbf)
{
  __shared__ __align__(16) unsigned short xs[16 * 32];
  const int t    = threadIdx.x;
  const int lane = t & 31;
  const int wave = t >> 5;
  const int mbase = blockIdx.y * 16;
  const int nbase = (blockIdx.x * 8 + wave) * 16;
  const int l15 = lane & 15;
  const int ab2 = (lane < 16) ? 0 : 8;
  const int bb2 = (lane < 16) ? 0 : 16;

  const unsigned short* W; int wrow;
  bool isq = false, isv = false;
  if (nbase < NQK)                 { W = wqbf; wrow = nbase;                  isq = true; }
  else if (nbase < NQK + HKV * HD) { W = wkbf; wrow = nbase - NQK; }
  else                             { W = wvbf; wrow = nbase - NQK - HKV*HD;  isv = true; }

  // staging: threads 0..127 move 4 bf16 (8B) each -> 16x32 bf16 tile
  const int srow = t >> 3;          // valid for t < 128
  const int scol = (t & 7) * 4;
  const unsigned short* gsrc = xbf + (size_t)(mbase + (srow & 15)) * DD + scol;
  unsigned short* ldst = &xs[(srow & 15) * 32 + scol];
  const unsigned short* wbase = W + (size_t)(wrow + l15) * DD;

  v8f acc = {};
  for (int kb = 0; kb < DD; kb += 32) {
    if (t < 128) {
#if HAS_ASYNC_LDS
      __builtin_amdgcn_global_load_async_to_lds_b64(
          (GLOBAL_AS v2i*)(gsrc + kb),
          (LDS_AS v2i*)ldst, 0, 0);
#else
      *(v2u*)ldst = *(const v2u*)(gsrc + kb);
#endif
    }
#if HAS_ASYNC_LDS
    wait_async0();
#endif
    if (kb + 32 < DD) __builtin_prefetch((const void*)(wbase + kb + 32), 0, 3);
    __syncthreads();

    BF16x16 af, bfr;
    const unsigned short* xr = &xs[l15 * 32 + ab2];
    af.q[0] = *(const v4u*)(xr);          // K = ab2..ab2+7
    af.q[1] = *(const v4u*)(xr + 16);     // K = 16+ab2..16+ab2+7
    const unsigned short* wr = wbase + kb + ab2;
    bfr.q[0] = *(const v4u*)(wr);         // K = ab2..ab2+7
    bfr.q[1] = *(const v4u*)(wr + 16);    // K = 16+ab2..16+ab2+7
    acc = wmma_bf16(af, bfr, acc);
    __syncthreads();
  }

  // Epilogue: RoPE (q/k only) in C-tile registers, write bf16 to workspace.
  const int half = (lane >> 4) * 8;
  #pragma unroll
  for (int r = 0; r < 8; r++) {
    float val = acc[r];
    int m = mbase + half + r;          // C row: M = r + 8*(lane/16)
    int b = m / SS, s = m % SS;
    int n = nbase + l15;               // C col: N = lane%16
    if (!isv) {
      int c = n & (HD - 1);
      int j = c >> 1;
      float cv = fcos[(size_t)s * (HD / 2) + j];
      float sv = fsin[(size_t)s * (HD / 2) + j];
      float partner = __shfl_xor(val, 1, 32);   // even<->odd column pair
      val = val * cv + ((c & 1) ? partner : -partner) * sv;
    }
    unsigned short bv = f2bf(val);
    if (isq) {
      int hh = n >> 6, c = n & 63;
      qbf[(((size_t)b * HH + hh) * SS + s) * HD + c] = bv;
    } else if (!isv) {
      int nk = n - NQK; int g = nk >> 6, c = nk & 63;
      kbf[(((size_t)b * HKV + g) * SS + s) * HD + c] = bv;
    } else {
      int nv = n - NQK - HKV * HD; int g = nv >> 6, c = nv & 63;
      vbf[(((size_t)b * HKV + g) * HD + c) * SS + s] = bv;   // transposed
    }
  }
}

// ---------------------------------------------------------------------------
// Online softmax + P*V for one 16-query tile (scores^T layout).
// MASKED instantiation only for causal-boundary chunks.
// ---------------------------------------------------------------------------
template <bool MASKED>
__device__ inline void softmax_pv(v8f& s0, v8f& s1, v8f* o, const BF16x16* vf,
                                  float& m_run, float& l_run,
                                  int kb, int qg, int half)
{
  float cmax = -1e30f;
  #pragma unroll
  for (int r = 0; r < 8; r++) {
    float sc0 = s0[r] * SCALEF;
    float sc1 = s1[r] * SCALEF;
    if (MASKED) {
      sc0 += ((kb + half + r)      > qg) ? NEGV : 0.0f;
      sc1 += ((kb + 16 + half + r) > qg) ? NEGV : 0.0f;
    }
    s0[r] = sc0; s1[r] = sc1;
    cmax = fmaxf(cmax, fmaxf(sc0, sc1));
  }
  cmax = fmaxf(cmax, __shfl_xor(cmax, 16, 32));
  float nm   = fmaxf(m_run, cmax);
  float corr = __expf(m_run - nm);
  float rs = 0.0f;
  #pragma unroll
  for (int r = 0; r < 8; r++) {
    s0[r] = __expf(s0[r] - nm);
    s1[r] = __expf(s1[r] - nm);
    rs += s0[r] + s1[r];
  }
  rs += __shfl_xor(rs, 16, 32);
  l_run = l_run * corr + rs;
  m_run = nm;

  float cr[8];
  #pragma unroll
  for (int r = 0; r < 8; r++) cr[r] = __shfl(corr, half + r, 32);
  #pragma unroll
  for (int dt = 0; dt < 4; dt++) {
    #pragma unroll
    for (int r = 0; r < 8; r++) o[dt][r] *= cr[r];
  }

  BF16x16 pf;   // exp'd score^T C-tile == A-fragment layout: pack row pairs
  #pragma unroll
  for (int v = 0; v < 4; v++) {
    pf.u[v]     = pack2f(s0[2 * v], s0[2 * v + 1]);
    pf.u[4 + v] = pack2f(s1[2 * v], s1[2 * v + 1]);
  }
  #pragma unroll
  for (int dt = 0; dt < 4; dt++) o[dt] = wmma_bf16(pf, vf[dt], o[dt]);
}

// ---------------------------------------------------------------------------
// Kernel 2: flash attention. One wave = 32 queries (two 16-q tiles sharing
// K and V fragments -> 16 WMMAs per 16 b128 loads per 32-key chunk).
// ---------------------------------------------------------------------------
__global__ __launch_bounds__(256) void attn_kernel(
    const unsigned short* __restrict__ qbf, const unsigned short* __restrict__ kbf,
    const unsigned short* __restrict__ vbf, unsigned short* __restrict__ abf)
{
  const int lane = threadIdx.x & 31;
  const int wave = threadIdx.x >> 5;
  const int qt = blockIdx.x * 8 + wave;
  const int h  = blockIdx.y;
  const int b  = blockIdx.z;
  const int g  = h / REP;
  const int qbase = qt * 32;
  const int l15 = lane & 15;
  const int half = (lane >> 4) * 8;
  const int ab2 = (lane < 16) ? 0 : 8;
  const int bb2 = (lane < 16) ? 0 : 16;

  const unsigned short* qp = qbf + ((size_t)(b * HH + h)  * SS) * HD;
  const unsigned short* kp = kbf + ((size_t)(b * HKV + g) * SS) * HD;
  const unsigned short* vp = vbf + ((size_t)(b * HKV + g) * HD) * SS; // [d][s]

  // Q^T B-fragments for both q-tiles, d-chunks [0,32) and [32,64)
  BF16x16 qa0, qa1, qb0, qb1;
  {
    const unsigned short* qr0 = qp + (size_t)(qbase + l15) * HD + ab2;
    const unsigned short* qr1 = qp + (size_t)(qbase + 16 + l15) * HD + ab2;
    qa0.q[0] = *(const v4u*)(qr0);      qa0.q[1] = *(const v4u*)(qr0 + 16);
    qa1.q[0] = *(const v4u*)(qr0 + 32); qa1.q[1] = *(const v4u*)(qr0 + 48);
    qb0.q[0] = *(const v4u*)(qr1);      qb0.q[1] = *(const v4u*)(qr1 + 16);
    qb1.q[0] = *(const v4u*)(qr1 + 32); qb1.q[1] = *(const v4u*)(qr1 + 48);
  }

  v8f oa[4] = {}, ob[4] = {};
  float m0 = -1e30f, l0 = 0.0f, m1 = -1e30f, l1 = 0.0f;
  const int qg0 = qbase + l15;
  const int qg1 = qbase + 16 + l15;

  for (int kb = 0; kb <= qbase + 31; kb += 32) {
    const bool do0 = (kb <= qbase + 15);   // tile0 still has unmasked keys
    {  // prefetch next 32-key chunk of K and V
      int pkb = (kb + 32 < SS) ? (kb + 32) : kb;
      __builtin_prefetch((const void*)(kp + (size_t)(pkb + lane) * HD), 0, 3);
      __builtin_prefetch((const void*)(vp + (size_t)(lane * 2)     * SS + pkb), 0, 3);
      __builtin_prefetch((const void*)(vp + (size_t)(lane * 2 + 1) * SS + pkb), 0, 3);
    }

    v8f sa0 = {}, sa1 = {}, sb0 = {}, sb1 = {};
    {  // keys [kb, kb+16): K A-fragments shared by both q-tiles
      const unsigned short* kr = kp + (size_t)(kb + l15) * HD + ab2;
      BF16x16 kf0, kf1;
      kf0.q[0] = *(const v4u*)(kr);      kf0.q[1] = *(const v4u*)(kr + 16);
      kf1.q[0] = *(const v4u*)(kr + 32); kf1.q[1] = *(const v4u*)(kr + 48);
      if (do0) {
        sa0 = wmma_bf16(kf0, qa0, sa0);
        sa0 = wmma_bf16(kf1, qa1, sa0);
      }
      sb0 = wmma_bf16(kf0, qb0, sb0);
      sb0 = wmma_bf16(kf1, qb1, sb0);
    }
    {  // keys [kb+16, kb+32)
      const unsigned short* kr = kp + (size_t)(kb + 16 + l15) * HD + ab2;
      BF16x16 kf0, kf1;
      kf0.q[0] = *(const v4u*)(kr);      kf0.q[1] = *(const v4u*)(kr + 16);
      kf1.q[0] = *(const v4u*)(kr + 32); kf1.q[1] = *(const v4u*)(kr + 48);
      if (do0) {
        sa1 = wmma_bf16(kf0, qa0, sa1);
        sa1 = wmma_bf16(kf1, qa1, sa1);
      }
      sb1 = wmma_bf16(kf0, qb0, sb1);
      sb1 = wmma_bf16(kf1, qb1, sb1);
    }

    // V B-fragments (transposed V): shared by both q-tiles
    BF16x16 vf[4];
    #pragma unroll
    for (int dt = 0; dt < 4; dt++) {
      const unsigned short* vr = vp + (size_t)(dt * 16 + l15) * SS + kb + ab2;
      vf[dt].q[0] = *(const v4u*)(vr);
      vf[dt].q[1] = *(const v4u*)(vr + 16);
    }

    if (do0) {
      if (kb + 31 > qbase)
        softmax_pv<true >(sa0, sa1, oa, vf, m0, l0, kb, qg0, half);
      else
        softmax_pv<false>(sa0, sa1, oa, vf, m0, l0, kb, qg0, half);
    }
    if (kb + 31 > qbase + 16)
      softmax_pv<true >(sb0, sb1, ob, vf, m1, l1, kb, qg1, half);
    else
      softmax_pv<false>(sb0, sb1, ob, vf, m1, l1, kb, qg1, half);
  }

  // normalize + store bf16 attn-out [B, S, H*HD]
  float linv0 = 1.0f / l0, linv1 = 1.0f / l1;
  float lr0[8], lr1[8];
  #pragma unroll
  for (int r = 0; r < 8; r++) {
    lr0[r] = __shfl(linv0, half + r, 32);
    lr1[r] = __shfl(linv1, half + r, 32);
  }
  #pragma unroll
  for (int r = 0; r < 8; r++) {
    int q0 = qbase + half + r;
    int q1 = qbase + 16 + half + r;
    size_t rb0 = ((size_t)b * SS + q0) * NQK + h * HD + l15;
    size_t rb1 = ((size_t)b * SS + q1) * NQK + h * HD + l15;
    #pragma unroll
    for (int dt = 0; dt < 4; dt++) {
      abf[rb0 + dt * 16] = f2bf(oa[dt][r] * lr0[r]);
      abf[rb1 + dt * 16] = f2bf(ob[dt][r] * lr1[r]);
    }
  }
}

// ---------------------------------------------------------------------------
// Kernel 3: output projection (attn bf16 [4096x1024]) x wo^T(bf16) -> f32 out
// ---------------------------------------------------------------------------
__global__ __launch_bounds__(256) void oproj_kernel(
    const unsigned short* __restrict__ abf, const unsigned short* __restrict__ wobf,
    float* __restrict__ out)
{
  const int lane = threadIdx.x & 31;
  const int wave = threadIdx.x >> 5;
  const int mbase = blockIdx.y * 16;
  const int nbase = (blockIdx.x * 8 + wave) * 16;
  const int l15 = lane & 15;
  const int ab2 = (lane < 16) ? 0 : 8;
  const int bb2 = (lane < 16) ? 0 : 16;

  const unsigned short* arow  = abf  + (size_t)(mbase + l15) * NQK;
  const unsigned short* wrowp = wobf + (size_t)(nbase + l15) * NQK;

  v8f acc = {};
  for (int kb = 0; kb < NQK; kb += 32) {
    if (kb + 32 < NQK) __builtin_prefetch((const void*)(wrowp + kb + 32), 0, 3);
    BF16x16 af, bfr;
    const unsigned short* ar = arow + kb + ab2;
    af.q[0]  = *(const v4u*)(ar);
    af.q[1]  = *(const v4u*)(ar + 16);
    const unsigned short* wr = wrowp + kb + ab2;
    bfr.q[0] = *(const v4u*)(wr);
    bfr.q[1] = *(const v4u*)(wr + 16);
    acc = wmma_bf16(af, bfr, acc);
  }
  const int half = (lane >> 4) * 8;
  #pragma unroll
  for (int r = 0; r < 8; r++)
    out[(size_t)(mbase + half + r) * DD + nbase + l15] = acc[r];
}

// ---------------------------------------------------------------------------
extern "C" void kernel_launch(void* const* d_in, const int* in_sizes, int n_in,
                              void* d_out, int out_size, void* d_ws, size_t ws_size,
                              hipStream_t stream) {
  (void)in_sizes; (void)n_in; (void)out_size; (void)ws_size;
  const float* x    = (const float*)d_in[0];
  const float* wq   = (const float*)d_in[1];
  const float* wk   = (const float*)d_in[2];
  const float* wv   = (const float*)d_in[3];
  const float* wo   = (const float*)d_in[4];
  const float* fcos = (const float*)d_in[5];
  const float* fsin = (const float*)d_in[6];
  // d_in[7] (mask) unused: causal mask computed analytically in-kernel.

  const size_t MB = (size_t)1 << 20;
  char* ws = (char*)d_ws;
  unsigned short* qbf  = (unsigned short*)(ws);                 // 8 MB
  unsigned short* kbf  = (unsigned short*)(ws + 8  * MB);       // 2 MB
  unsigned short* vbf  = (unsigned short*)(ws + 10 * MB);       // 2 MB (transposed)
  unsigned short* abf  = (unsigned short*)(ws + 12 * MB);       // 8 MB
  unsigned short* xbf  = (unsigned short*)(ws + 20 * MB);       // 8 MB
  unsigned short* wqbf = (unsigned short*)(ws + 28 * MB);       // 2 MB
  unsigned short* wkbf = (unsigned short*)(ws + 30 * MB);       // 0.5 MB
  unsigned short* wvbf = (unsigned short*)(ws + 30 * MB + 512 * 1024); // 0.5 MB
  unsigned short* wobf = (unsigned short*)(ws + 31 * MB);       // 2 MB

  const int XN = BB * SS * DD;      // 4,194,304
  const int WQN = NQK * DD;         // 1,048,576
  const int WKN = HKV * HD * DD;    //   262,144
  cvt_bf16_kernel<<<(XN / 4 + 255) / 256, 256, 0, stream>>>(x,  xbf,  XN / 4);
  cvt_bf16_kernel<<<(WQN / 4 + 255) / 256, 256, 0, stream>>>(wq, wqbf, WQN / 4);
  cvt_bf16_kernel<<<(WKN / 4 + 255) / 256, 256, 0, stream>>>(wk, wkbf, WKN / 4);
  cvt_bf16_kernel<<<(WKN / 4 + 255) / 256, 256, 0, stream>>>(wv, wvbf, WKN / 4);
  cvt_bf16_kernel<<<(WQN / 4 + 255) / 256, 256, 0, stream>>>(wo, wobf, WQN / 4);

  dim3 g1(NTOT / 16 / 8, (BB * SS) / 16);   // (12, 256)
  qkv_rope_kernel<<<g1, 256, 0, stream>>>(xbf, wqbf, wkbf, wvbf, fcos, fsin,
                                          qbf, kbf, vbf);

  dim3 g2(SS / 32 / 8, HH, BB);             // (8, 16, 2)
  attn_kernel<<<g2, 256, 0, stream>>>(qbf, kbf, vbf, abf);

  dim3 g3(DD / 16 / 8, (BB * SS) / 16);     // (8, 256)
  oproj_kernel<<<g3, 256, 0, stream>>>(abf, wobf, (float*)d_out);
}
